// ArDiffusion_50337016709884
// MI455X (gfx1250) — matrix operand order repaired and checked by reading, not compiled, not run on hardware
//
#include <hip/hip_runtime.h>
#include <hip/hip_bf16.h>
#include <math.h>

// ---------------- CDNA5 WMMA types ----------------
typedef __attribute__((ext_vector_type(16))) __bf16 v16bf;
typedef __attribute__((ext_vector_type(8)))  __bf16 v8bf;
typedef __attribute__((ext_vector_type(8)))  float  v8f;

// ---------------- problem constants ----------------
#define VOCAB  50257
#define VPAD   50272            // 3142 * 16
#define NT2    (VPAD / 16)      // 3142 n-tiles for GEMM2
#define NBY    ((NT2 + 7) / 8)  // 393 column-blocks of GEMM2
#define NSd    8
#define NEd    1024
#define EPSd   128
#define Bd     4
#define Td     2048
#define Ld     2055             // tilted sequence length
#define MROWS  (Bd * Ld)        // 8220
#define MPAD   8224             // 257 * 32
#define M2     (Bd * (Td - 1))  // 8188 logits rows
#define M2PAD  8192
#define LNEPS  1e-5f
#define NEGINF (-3.4e38f)

// ---------------- WMMA helpers (ISA 7.12.2 layouts, wave32) ----------------
static __device__ __forceinline__ v8f wmma_bf16(v16bf a, v16bf b, v8f c) {
  // v_wmma_f32_16x16x32_bf16
  return __builtin_amdgcn_wmma_f32_16x16x32_bf16(false, a, false, b, (short)0, c,
                                                 false, false);
}

// A fragment (16x32 bf16, row-major src, ld in elements):
// lane m = lane&15, h = lane>>4 ; element i: k = kbase + i + 16*h
static __device__ __forceinline__ v16bf load_a_frag(const __bf16* p /* = &A[m*ld + kbase + 16*h] */) {
  v8bf lo = *(const v8bf*)(p);        // k = kbase+16h .. +7
  v8bf hi = *(const v8bf*)(p + 8);    // k = kbase+16h+8 .. +7
  v16bf f;
#pragma unroll
  for (int i = 0; i < 8; ++i) { f[i] = lo[i]; f[i + 8] = hi[i]; }
  return f;
}

// B fragment (32x16 bf16) loaded from an n-major (transposed) buffer:
// lane n = lane&15, h = lane>>4 ; element i: k = kbase + i + 16*h  (32 contiguous bytes)
static __device__ __forceinline__ v16bf load_b_fragT(const __bf16* p /* = &BT[n*ld + kbase + 16*h] */) {
  return *(const v16bf*)p;
}

// ---------------- conversion kernels ----------------
__global__ void cvt_wbT_kernel(const float* __restrict__ Wb, __bf16* __restrict__ wbT) {
  int idx = blockIdx.x * 256 + threadIdx.x;      // over NEd*NEd
  if (idx >= NEd * NEd) return;
  int n = idx >> 10, k = idx & 1023;             // contiguous writes in k
  wbT[idx] = (__bf16)Wb[k * NEd + n];
}

__global__ void cvt_wte_kernel(const float* __restrict__ wte, __bf16* __restrict__ wteb) {
  int idx = blockIdx.x * 256 + threadIdx.x;      // over VPAD*EPSd
  if (idx >= VPAD * EPSd) return;
  int v = idx >> 7, e = idx & 127;
  wteb[idx] = (v < VOCAB) ? (__bf16)wte[(size_t)v * EPSd + e] : (__bf16)0.0f;
}

__global__ void zero_y7pad_kernel(__bf16* __restrict__ y7) {
  int idx = blockIdx.x * 256 + threadIdx.x;      // (M2PAD-M2)*EPSd = 512
  if (idx < (M2PAD - M2) * EPSd) y7[(size_t)M2 * EPSd + idx] = (__bf16)0.0f;
}

// ---------------- stage A: fused gather / mix / tilt / +wpe ----------------
__global__ void build_x_kernel(const int* __restrict__ toks, const float* __restrict__ wte,
                               const float* __restrict__ wpe, const float* __restrict__ lno,
                               const float* __restrict__ rno, const float* __restrict__ noi,
                               float* __restrict__ xin, __bf16* __restrict__ xbf) {
  int idx = blockIdx.x * 256 + threadIdx.x;      // over MPAD*NEd
  if (idx >= MPAD * NEd) return;
  int ne = idx & 1023, row = idx >> 10;
  float v = 0.f, xf = 0.f;
  if (row < MROWS) {
    int b = row / Ld, l = row % Ld;
    int s = ne >> 7, e = ne & 127;
    int j = l + s;                               // position in concat axis (len 2062)
    if (j < NSd - 1) {
      v = lno[((b * 7 + j) * NSd + s) * EPSd + e];
    } else if (j >= NSd - 1 + Td) {
      v = rno[((b * 7 + (j - (NSd - 1 + Td))) * NSd + s) * EPSd + e];
    } else {
      int t = j - (NSd - 1);
      int tok = toks[b * Td + t];
      float w = (float)s * 0.125f;
      v = wte[(size_t)tok * EPSd + e] * (1.f - w) +
          noi[((size_t)(b * Td + t)) * EPSd + e] * w;
    }
    xf = v + wpe[l * NEd + ne];
  }
  xin[idx] = v;                 // x_in (no wpe) for latent loss
  xbf[idx] = (__bf16)xf;        // x_flat in bf16 for GEMM1
}

// ---------------- GEMM1: y = x_flat @ Wb  (8220x1024 @ 1024x1024) ----------------
__global__ void __launch_bounds__(256) gemm1_kernel(const __bf16* __restrict__ X,
                                                    const __bf16* __restrict__ WT,
                                                    float* __restrict__ Y) {
  const int lane = threadIdx.x & 31;
  const int wave = threadIdx.x >> 5;
  const int m0 = blockIdx.x * 32;                  // two 16-row tiles
  const int n0 = blockIdx.y * 256 + wave * 32;     // two 16-col tiles
  const int mr = lane & 15, h = lane >> 4;

  v8f c00 = {}, c01 = {}, c10 = {}, c11 = {};
  const __bf16* a0p = X + (size_t)(m0 + mr) * NEd + 16 * h;
  const __bf16* a1p = X + (size_t)(m0 + 16 + mr) * NEd + 16 * h;
  const __bf16* b0p = WT + (size_t)(n0 + mr) * NEd + 16 * h;
  const __bf16* b1p = WT + (size_t)(n0 + 16 + mr) * NEd + 16 * h;

  for (int k = 0; k < NEd; k += 32) {
    v16bf a0 = load_a_frag(a0p + k);
    v16bf a1 = load_a_frag(a1p + k);
    v16bf b0 = load_b_fragT(b0p + k);
    v16bf b1 = load_b_fragT(b1p + k);
    c00 = wmma_bf16(a0, b0, c00);
    c01 = wmma_bf16(a0, b1, c01);
    c10 = wmma_bf16(a1, b0, c10);
    c11 = wmma_bf16(a1, b1, c11);
  }

  // C/D layout: element r -> m = r + 8*h, n = lane&15
  const int nA = n0 + mr, nB = n0 + 16 + mr;
  const int mbase0 = m0 + 8 * h, mbase1 = m0 + 16 + 8 * h;
#pragma unroll
  for (int r = 0; r < 8; ++r) {
    int mA = mbase0 + r, mB = mbase1 + r;
    if (mA < MROWS) { Y[(size_t)mA * NEd + nA] = c00[r]; Y[(size_t)mA * NEd + nB] = c01[r]; }
    if (mB < MROWS) { Y[(size_t)mB * NEd + nA] = c10[r]; Y[(size_t)mB * NEd + nB] = c11[r]; }
  }
}

// ---------------- LayerNorm (in place) + bf16 slice for GEMM2 ----------------
__global__ void __launch_bounds__(256) ln_kernel(const float* __restrict__ ln_g,
                                                 float* __restrict__ Y,
                                                 __bf16* __restrict__ Y7) {
  __shared__ float red[256];
  const int row = blockIdx.x;                    // 0..MROWS-1
  const int tid = threadIdx.x;
  float* y = Y + (size_t)row * NEd;

  float x0 = y[tid], x1 = y[tid + 256], x2 = y[tid + 512], x3 = y[tid + 768];
  red[tid] = x0 + x1 + x2 + x3;
  __syncthreads();
  for (int o = 128; o > 0; o >>= 1) { if (tid < o) red[tid] += red[tid + o]; __syncthreads(); }
  const float mu = red[0] * (1.f / NEd);
  __syncthreads();

  float d0 = x0 - mu, d1 = x1 - mu, d2 = x2 - mu, d3 = x3 - mu;
  red[tid] = d0 * d0 + d1 * d1 + d2 * d2 + d3 * d3;
  __syncthreads();
  for (int o = 128; o > 0; o >>= 1) { if (tid < o) red[tid] += red[tid + o]; __syncthreads(); }
  const float rstd = rsqrtf(red[0] * (1.f / NEd) + LNEPS);

  const int b = row / Ld, l = row % Ld;
  float d[4] = {d0, d1, d2, d3};
#pragma unroll
  for (int i = 0; i < 4; ++i) {
    int ne = tid + 256 * i;
    float yn = d[i] * rstd * ln_g[ne];
    y[ne] = yn;
    if (l >= NSd && ne >= (NSd - 1) * EPSd) {     // s == 7 slice, rows l in [8,2055)
      Y7[((size_t)(b * (Td - 1) + (l - NSd))) * EPSd + (ne - (NSd - 1) * EPSd)] = (__bf16)yn;
    }
  }
}

// ---- GEMM2 (fused): logits = y7 @ wte^T, plus per-(row, col-block) logsumexp partials ----
__global__ void __launch_bounds__(256) gemm2_kernel(const __bf16* __restrict__ Y7,
                                                    const __bf16* __restrict__ WTEB,
                                                    float* __restrict__ out,
                                                    float* __restrict__ pmax,
                                                    float* __restrict__ psum) {
  const int lane = threadIdx.x & 31;
  const int wave = threadIdx.x >> 5;
  const int m0 = blockIdx.x * 16;
  const int nt = blockIdx.y * 8 + wave;          // wave-uniform activity predicate
  const int mr = lane & 15, h = lane >> 4;

  __shared__ float smax[16][8];
  __shared__ float ssum[16][8];

  float rowmax[8], rowsum[8];
  const bool active = (nt < NT2);
  if (active) {                                  // wave-uniform: EXEC all-ones for WMMA
    const int n0 = nt * 16;
    const __bf16* ap = Y7 + (size_t)(m0 + mr) * EPSd + 16 * h;
    const __bf16* bp = WTEB + (size_t)(n0 + mr) * EPSd + 16 * h;

    v8f c = {};
#pragma unroll
    for (int kk = 0; kk < 4; ++kk) {             // K = 128 = 4 * 32
      v16bf a = load_a_frag(ap + kk * 32);
      v16bf b = load_b_fragT(bp + kk * 32);
      c = wmma_bf16(a, b, c);
    }

    // store logits + build masked values for the softmax partials
    const int n = n0 + mr;
    const bool nvalid = (n < VOCAB);
    float vals[8];
#pragma unroll
    for (int r = 0; r < 8; ++r) {
      int m = m0 + 8 * h + r;
      vals[r] = nvalid ? c[r] : NEGINF;
      if (nvalid && m < M2) out[(size_t)m * VOCAB + n] = c[r];
    }

    // butterfly max over the 16 lanes holding this tile's 16 columns
#pragma unroll
    for (int r = 0; r < 8; ++r) rowmax[r] = vals[r];
    for (int mask = 1; mask < 16; mask <<= 1) {
#pragma unroll
      for (int r = 0; r < 8; ++r)
        rowmax[r] = fmaxf(rowmax[r], __shfl_xor(rowmax[r], mask, 32));
    }
    // butterfly sum of exp(val - rowmax)
#pragma unroll
    for (int r = 0; r < 8; ++r) rowsum[r] = expf(vals[r] - rowmax[r]);
    for (int mask = 1; mask < 16; mask <<= 1) {
#pragma unroll
      for (int r = 0; r < 8; ++r)
        rowsum[r] += __shfl_xor(rowsum[r], mask, 32);
    }
  } else {
#pragma unroll
    for (int r = 0; r < 8; ++r) { rowmax[r] = NEGINF; rowsum[r] = 0.f; }
  }

  // one lane per half-wave publishes its 8 rows
  if (mr == 0) {
#pragma unroll
    for (int r = 0; r < 8; ++r) {
      smax[8 * h + r][wave] = rowmax[r];
      ssum[8 * h + r][wave] = rowsum[r];
    }
  }
  __syncthreads();

  // wave 0 merges the 8 waves (128 columns) per row and writes block partials
  if (wave == 0 && lane < 16) {
    float M = NEGINF;
#pragma unroll
    for (int w2 = 0; w2 < 8; ++w2) M = fmaxf(M, smax[lane][w2]);
    float S = 0.f;
#pragma unroll
    for (int w2 = 0; w2 < 8; ++w2) S += ssum[lane][w2] * expf(smax[lane][w2] - M);
    pmax[(size_t)(m0 + lane) * NBY + blockIdx.y] = M;
    psum[(size_t)(m0 + lane) * NBY + blockIdx.y] = S;
  }
}

// ---------------- CE per row from GEMM2 partials (online logsumexp merge) ----------------
__global__ void __launch_bounds__(256) ce_combine_kernel(const float* __restrict__ pmax,
                                                         const float* __restrict__ psum,
                                                         const float* __restrict__ logits,
                                                         const int* __restrict__ toks,
                                                         float* __restrict__ ce_rows) {
  __shared__ float rm[256], rs[256];
  const int row = blockIdx.x;                    // 0..M2-1
  const int tid = threadIdx.x;

  float M = NEGINF, S = 0.f;
  for (int i = tid; i < NBY; i += 256) {
    float m2 = pmax[(size_t)row * NBY + i];
    float s2 = psum[(size_t)row * NBY + i];
    if (m2 > M) { S = S * expf(M - m2) + s2; M = m2; }
    else        { S += s2 * expf(m2 - M); }
  }
  rm[tid] = M; rs[tid] = S;
  __syncthreads();
  for (int o = 128; o > 0; o >>= 1) {
    if (tid < o) {
      float M1 = rm[tid], S1 = rs[tid];
      float M2v = rm[tid + o], S2v = rs[tid + o];
      float Mn = fmaxf(M1, M2v);
      rs[tid] = S1 * expf(M1 - Mn) + S2v * expf(M2v - Mn);
      rm[tid] = Mn;
    }
    __syncthreads();
  }
  if (tid == 0) {
    int b = row / (Td - 1), i = row % (Td - 1);
    int tgt = toks[b * Td + i + 1];
    ce_rows[row] = rm[0] + logf(rs[0]) - logits[(size_t)row * VOCAB + tgt];
  }
}

// ---------------- latent partial sums ----------------
#define NLATB 1024
__global__ void __launch_bounds__(256) latent_kernel(const float* __restrict__ Y,
                                                     const float* __restrict__ XIN,
                                                     float* __restrict__ part) {
  __shared__ float red[256];
  const size_t total = (size_t)Bd * (Ld - 1) * NEd;   // 4*2054*1024
  float acc = 0.f;
  for (size_t idx = (size_t)blockIdx.x * 256 + threadIdx.x; idx < total;
       idx += (size_t)gridDim.x * 256) {
    int ne = (int)(idx & 1023);
    size_t rr = idx >> 10;
    int l = (int)(rr % (Ld - 1));
    int b = (int)(rr / (Ld - 1));
    int s = ne >> 7;
    int lp = l + 1;
    if (lp + s >= NSd - 1 && lp + s <= Ld - 1) {      // mask: 7 <= l'+s <= 2054
      float yv = Y[((size_t)(b * Ld + l)) * NEd + ne];
      float xv = XIN[((size_t)(b * Ld + lp)) * NEd + ne];
      float d = yv - xv;
      acc += d * d;
    }
  }
  red[threadIdx.x] = acc;
  __syncthreads();
  for (int o = 128; o > 0; o >>= 1) { if (threadIdx.x < o) red[threadIdx.x] += red[threadIdx.x + o]; __syncthreads(); }
  if (threadIdx.x == 0) part[blockIdx.x] = red[0];
}

// ---------------- final scalar: loss = ce_mean + latent ----------------
__global__ void __launch_bounds__(256) final_kernel(const float* __restrict__ ce_rows,
                                                    const float* __restrict__ part,
                                                    float* __restrict__ loss_out) {
  __shared__ float red[256];
  const int tid = threadIdx.x;

  float a = 0.f;
  for (int i = tid; i < M2; i += 256) a += ce_rows[i];
  red[tid] = a;
  __syncthreads();
  for (int o = 128; o > 0; o >>= 1) { if (tid < o) red[tid] += red[tid + o]; __syncthreads(); }
  const float ce_sum = red[0];
  __syncthreads();

  float b = 0.f;
  for (int i = tid; i < NLATB; i += 256) b += part[i];
  red[tid] = b;
  __syncthreads();
  for (int o = 128; o > 0; o >>= 1) { if (tid < o) red[tid] += red[tid + o]; __syncthreads(); }

  if (tid == 0) {
    // mask element count: broadcast over b (4) and e (128)
    long cnt = 0;
    for (int lp = 1; lp <= Ld - 1; ++lp)
      for (int s = 0; s < NSd; ++s)
        if (lp + s >= NSd - 1 && lp + s <= Ld - 1) ++cnt;
    float msum = (float)cnt * (float)Bd * (float)EPSd;
    loss_out[0] = ce_sum / (float)M2 + red[0] / msum;
  }
}

// ---------------- host-side launch ----------------
extern "C" void kernel_launch(void* const* d_in, const int* in_sizes, int n_in,
                              void* d_out, int out_size, void* d_ws, size_t ws_size,
                              hipStream_t stream) {
  const int*   toks = (const int*)  d_in[0];
  const float* wte  = (const float*)d_in[1];
  const float* wpe  = (const float*)d_in[2];
  const float* Wb   = (const float*)d_in[3];
  const float* ln_g = (const float*)d_in[4];
  const float* lno  = (const float*)d_in[5];
  const float* rno  = (const float*)d_in[6];
  const float* noi  = (const float*)d_in[7];
  float* out = (float*)d_out;

  // workspace carve-up (each chunk 256B aligned)
  char* w = (char*)d_ws;
  size_t off = 0;
  auto take = [&](size_t bytes) -> void* {
    void* p = w + off;
    off += (bytes + 255) & ~(size_t)255;
    return p;
  };
  float*  xin  = (float*) take((size_t)MPAD * NEd * 4);
  __bf16* xbf  = (__bf16*)take((size_t)MPAD * NEd * 2);
  __bf16* wbT  = (__bf16*)take((size_t)NEd * NEd * 2);
  float*  y    = (float*) take((size_t)MPAD * NEd * 4);
  __bf16* wteb = (__bf16*)take((size_t)VPAD * EPSd * 2);
  __bf16* y7   = (__bf16*)take((size_t)M2PAD * EPSd * 2);
  float*  pmax = (float*) take((size_t)M2PAD * NBY * 4);
  float*  psum = (float*) take((size_t)M2PAD * NBY * 4);
  float*  cer  = (float*) take((size_t)M2 * 4);
  float*  lpart= (float*) take((size_t)NLATB * 4);
  (void)ws_size; (void)in_sizes; (void)n_in; (void)out_size;

  cvt_wbT_kernel <<<(NEd * NEd) / 256, 256, 0, stream>>>(Wb, wbT);
  cvt_wte_kernel <<<(VPAD * EPSd) / 256, 256, 0, stream>>>(wte, wteb);
  zero_y7pad_kernel<<<2, 256, 0, stream>>>(y7);
  build_x_kernel <<<(MPAD * NEd) / 256, 256, 0, stream>>>(toks, wte, wpe, lno, rno, noi, xin, xbf);
  gemm1_kernel   <<<dim3(MPAD / 32, NEd / 256), 256, 0, stream>>>(xbf, wbT, y);
  ln_kernel      <<<MROWS, 256, 0, stream>>>(ln_g, y, y7);
  gemm2_kernel   <<<dim3(M2PAD / 16, NBY), 256, 0, stream>>>(y7, wteb, out, pmax, psum);
  ce_combine_kernel<<<M2, 256, 0, stream>>>(pmax, psum, out, toks, cer);
  latent_kernel  <<<NLATB, 256, 0, stream>>>(y, xin, lpart);
  final_kernel   <<<1, 256, 0, stream>>>(cer, lpart, out + (size_t)M2 * VOCAB);
}
